// Attention_45947560133325
// MI455X (gfx1250) — matrix operand order repaired and checked by reading, not compiled, not run on hardware
//
#include <hip/hip_runtime.h>

// ---------------------------------------------------------------------------
// MI455X (gfx1250) fused attention: bf16 WMMA (16x16x32) with f32 accumulate.
// Pipeline:
//   [cvt f32->bf16 prepass for x/qkv_w/proj_w]
//   [QKV gemm, 64x64 wave tile -> bf16 ws]
//   [flash attention, S^T/O^T trick]
//   [proj gemm, 64x64 wave tile]
// ---------------------------------------------------------------------------

typedef __attribute__((ext_vector_type(16))) __bf16 v16bf;
typedef __attribute__((ext_vector_type(8)))  float  v8f;

union BF16Frag {
  v16bf b;
  __bf16 e[16];
  uint4 q[2];
};

union Pack8 {
  __bf16 e[8];
  uint4 q;
};

__device__ __forceinline__ __bf16 f2bf(float f) {
  return (__bf16)f;  // hardware cvt path
}

__device__ __forceinline__ v8f wmma_bf16(v16bf a, v16bf b, v8f c) {
  // D = A(16x32 bf16) x B(32x16 bf16) + C(16x16 f32)
  return __builtin_amdgcn_wmma_f32_16x16x32_bf16(
      /*neg_a=*/false, a, /*neg_b=*/false, b,
      /*c_mod=*/(short)0, c, /*reuse_a=*/false, /*reuse_b=*/false);
}

// A-fragment (16x32 bf16). Caller pre-offsets p by (8*hi) along K.
// elements 0..7  <- p[0..7]   (K = 8*hi + 0..7)
// elements 8..15 <- p[16..23] (K = 8*hi + 16..23)
__device__ __forceinline__ v16bf frag_a_bf16(const __bf16* __restrict__ p) {
  BF16Frag fr;
  fr.q[0] = *(const uint4*)(p);
  fr.q[1] = *(const uint4*)(p + 16);
  return fr.b;
}

// B-fragment (32x16 bf16). Caller pre-offsets p by (8*hi) along K.
// elements 0..7  <- p[0..7]   (K = 8*hi + 0..7)
// elements 8..15 <- p[16..23] (K = 8*hi + 16..23)
__device__ __forceinline__ v16bf frag_b_bf16(const __bf16* __restrict__ p) {
  BF16Frag fr;
  fr.q[0] = *(const uint4*)(p);
  fr.q[1] = *(const uint4*)(p + 16);
  return fr.b;
}

// Problem constants
#define CDIM  768
#define NTOK  512
#define HEADS 12
#define DH    64
#define BT    16          // B*T
#define BTH   (BT * HEADS)
#define MROWS (BT * NTOK) // 8192

// ---------------------------------------------------------------------------
// Kernel 0: elementwise f32 -> bf16 (8 elems / thread, fully coalesced).
// ---------------------------------------------------------------------------
__global__ __launch_bounds__(256) void cvt_kernel(
    const float* __restrict__ src, __bf16* __restrict__ dst, int n8) {
  const int i = blockIdx.x * 256 + threadIdx.x;
  if (i >= n8) return;
  const size_t off = (size_t)i * 8;
  float4 a = *(const float4*)(src + off);
  float4 b = *(const float4*)(src + off + 4);
  Pack8 u;
  u.e[0] = f2bf(a.x); u.e[1] = f2bf(a.y); u.e[2] = f2bf(a.z); u.e[3] = f2bf(a.w);
  u.e[4] = f2bf(b.x); u.e[5] = f2bf(b.y); u.e[6] = f2bf(b.z); u.e[7] = f2bf(b.w);
  *(uint4*)(dst + off) = u.q;
}

// ---------------------------------------------------------------------------
// Kernel 1: qkv = x @ qkv_w^T + cat(q_bias, 0, v_bias), scattered to
//   qws,kws : bf16 [bth][n][d]     vT : bf16 [bth][d][n]
// grid (32, 36), block 128 (4 waves), wave tile = 64(M) x 64(N):
//   4 A-frags x 4 B-frags -> 16 WMMAs per K-step, 1.0 b128-load / WMMA.
// Epilogue routing (q/k/v, head) is block-uniform: n-slab = 64 cols, 64 | 768.
// ---------------------------------------------------------------------------
__global__ __launch_bounds__(128) void qkv_kernel(
    const __bf16* __restrict__ xb, const __bf16* __restrict__ wb,
    const float* __restrict__ q_bias, const float* __restrict__ v_bias,
    __bf16* __restrict__ qws, __bf16* __restrict__ kws,
    __bf16* __restrict__ vT) {
  const int lane = threadIdx.x & 31;
  const int wave = threadIdx.x >> 5;
  const int hi   = lane >> 4;
  const int lid  = lane & 15;

  const int m0 = blockIdx.x * 256 + wave * 64;
  const int n0 = blockIdx.y * 64;

  const __bf16* xrow[4];
#pragma unroll
  for (int a = 0; a < 4; ++a)
    xrow[a] = xb + (size_t)(m0 + a * 16 + lid) * CDIM;

  const v8f vzero = {0.f, 0.f, 0.f, 0.f, 0.f, 0.f, 0.f, 0.f};
  v8f acc[4][4];
#pragma unroll
  for (int a = 0; a < 4; ++a)
#pragma unroll
    for (int j = 0; j < 4; ++j) acc[a][j] = vzero;

  for (int k = 0; k < CDIM; k += 32) {
    v16bf af[4];
#pragma unroll
    for (int a = 0; a < 4; ++a) af[a] = frag_a_bf16(xrow[a] + k + 8 * hi);
#pragma unroll
    for (int j = 0; j < 4; ++j) {
      const __bf16* wrow = wb + (size_t)(n0 + j * 16 + lid) * CDIM;
      const v16bf b = frag_b_bf16(wrow + k + 8 * hi);
#pragma unroll
      for (int a = 0; a < 4; ++a) acc[a][j] = wmma_bf16(af[a], b, acc[a][j]);
    }
  }

  // ---- epilogue: all routing decisions are scalar (blockIdx-derived) ----
  const int bt    = m0 >> 9;     // 64-row wave tile never straddles a token blk
  const int byy   = blockIdx.y;
  const int which = (byy >= 24) ? 2 : (byy >= 12) ? 1 : 0;  // 0:q 1:k 2:v
  const int nrem  = (byy - which * 12) * 64;   // column offset inside q/k/v
  const int hh    = nrem >> 6;                 // head index (uniform: 64-col slab)
  const int bth   = bt * HEADS + hh;

#pragma unroll
  for (int a = 0; a < 4; ++a) {
    const int tok0 = (m0 & 511) + a * 16 + 8 * hi;   // D rows: tok0 + r
#pragma unroll
    for (int j = 0; j < 4; ++j) {
      const int dh = j * 16 + lid;                   // dim-in-head (n0 & 63 == 0)
      const float bias = (which == 0) ? q_bias[nrem + dh]
                       : (which == 2) ? v_bias[nrem + dh] : 0.f;
      if (which == 2) {
        // V stored transposed: [bth][dh][tok]; 8 consecutive tokens -> one uint4
        Pack8 u;
#pragma unroll
        for (int r = 0; r < 8; ++r) u.e[r] = f2bf(acc[a][j][r] + bias);
        *(uint4*)(vT + ((size_t)bth * DH + dh) * NTOK + tok0) = u.q;
      } else {
        __bf16* dst = (which == 0) ? qws : kws;
#pragma unroll
        for (int r = 0; r < 8; ++r)
          dst[((size_t)bth * NTOK + tok0 + r) * DH + dh] = f2bf(acc[a][j][r] + bias);
      }
    }
  }
}

// ---------------------------------------------------------------------------
// Kernel 2: flash attention per head.
//   S^T = K Q^T  (query in lane dim -> lane-uniform softmax stats)
//   O^T = V^T P^T (rescale factor also lane-uniform)
// grid (192, 8), block 128 (4 waves); each wave owns 16 queries.
// ---------------------------------------------------------------------------
__global__ __launch_bounds__(128) void attn_kernel(
    const __bf16* __restrict__ qws, const __bf16* __restrict__ kws,
    const __bf16* __restrict__ vT, const float* __restrict__ mask,
    __bf16* __restrict__ aout) {
  const int lane = threadIdx.x & 31;
  const int wave = threadIdx.x >> 5;
  const int hi   = lane >> 4;
  const int lid  = lane & 15;

  const int bth = blockIdx.x;
  const int bt  = bth / HEADS;
  const int h   = bth % HEADS;
  const int q   = blockIdx.y * 64 + wave * 16 + lid;   // this lane's query token

  // Q^T B-fragments (fixed for the whole key loop); d-chunks 0..31 and 32..63
  const __bf16* qrow = qws + ((size_t)bth * NTOK + q) * DH;
  const v16bf bq0 = frag_b_bf16(qrow + 8 * hi);
  const v16bf bq1 = frag_b_bf16(qrow + 32 + 8 * hi);

  const float scale = 0.125f;  // Dh^-0.5
  float m_run = -3.0e38f, l_run = 0.f;
  const v8f vzero = {0.f, 0.f, 0.f, 0.f, 0.f, 0.f, 0.f, 0.f};
  v8f o[4];
#pragma unroll
  for (int f = 0; f < 4; ++f) o[f] = vzero;

  const float* mrowbase = mask + (size_t)q * NTOK;

  for (int kb = 0; kb < NTOK; kb += 32) {
    // ---- S^T tiles: t=0 -> keys kb..kb+15, t=1 -> keys kb+16..kb+31
    float sarr[2][8];
    float mloc = -3.0e38f;
#pragma unroll
    for (int t = 0; t < 2; ++t) {
      const int key = kb + t * 16 + lid;
      const __bf16* krow = kws + ((size_t)bth * NTOK + key) * DH;
      v8f c = vzero;
      c = wmma_bf16(frag_a_bf16(krow + 8 * hi), bq0, c);
      c = wmma_bf16(frag_a_bf16(krow + 32 + 8 * hi), bq1, c);
      // keys this lane holds: kb + t*16 + 8*hi + r, r = 0..7
      const float* mrow = mrowbase + kb + t * 16 + 8 * hi;
      float4 ma = *(const float4*)(mrow);
      float4 mb = *(const float4*)(mrow + 4);
      float mv[8] = {ma.x, ma.y, ma.z, ma.w, mb.x, mb.y, mb.z, mb.w};
#pragma unroll
      for (int r = 0; r < 8; ++r) {
        float vsc = c[r] * scale + mv[r];
        sarr[t][r] = vsc;
        mloc = fmaxf(mloc, vsc);
      }
    }
    // ---- online softmax (stats for query `lid`, replicated on lane pair)
    mloc = fmaxf(mloc, __shfl_xor(mloc, 16, 32));
    const float m_new = fmaxf(m_run, mloc);
    const float sf = __expf(m_run - m_new);
    float p0[8], p1[8], ssum = 0.f;
#pragma unroll
    for (int r = 0; r < 8; ++r) {
      p0[r] = __expf(sarr[0][r] - m_new);
      p1[r] = __expf(sarr[1][r] - m_new);
      ssum += p0[r] + p1[r];
    }
    ssum += __shfl_xor(ssum, 16, 32);
    l_run = l_run * sf + ssum;
    m_run = m_new;

    // ---- build P^T B-fragment: B wants K = 8*hi + r (elems 0..7) and
    // 16 + 8*hi + r (elems 8..15); our C-tiles hold exactly those keys.
    BF16Frag pf;
#pragma unroll
    for (int r = 0; r < 8; ++r) {
      pf.e[r] = f2bf(p0[r]);
      pf.e[8 + r] = f2bf(p1[r]);
    }
    const v16bf pT = pf.b;

    // ---- O^T += V^T x P^T  (4 tiles over Dh); rescale is lane-uniform
#pragma unroll
    for (int f = 0; f < 4; ++f) {
      const int dh = f * 16 + lid;
      const __bf16* vrow = vT + ((size_t)bth * DH + dh) * NTOK + kb;
      v16bf av = frag_a_bf16(vrow + 8 * hi);
      v8f osc = o[f] * sf;
      o[f] = wmma_bf16(av, pT, osc);
    }
  }

  // ---- normalize and store O^T -> aout bf16 [row=bt*512+q][h*64+dh]
  const float inv = 1.0f / l_run;
  const size_t outbase = ((size_t)bt * NTOK + q) * CDIM + h * DH;
#pragma unroll
  for (int f = 0; f < 4; ++f) {
    Pack8 u;
#pragma unroll
    for (int r = 0; r < 8; ++r) u.e[r] = f2bf(o[f][r] * inv);
    *(uint4*)(aout + outbase + f * 16 + 8 * hi) = u.q;  // dh = f*16 + 8*hi + r
  }
}

// ---------------------------------------------------------------------------
// Kernel 3: out = attn_out @ proj_w^T + proj_b   (fp32 output)
// grid (32, 12), block 128; wave tile 64(M) x 64(N)
// ---------------------------------------------------------------------------
__global__ __launch_bounds__(128) void proj_kernel(
    const __bf16* __restrict__ aout, const __bf16* __restrict__ wb,
    const float* __restrict__ proj_b, float* __restrict__ out) {
  const int lane = threadIdx.x & 31;
  const int wave = threadIdx.x >> 5;
  const int hi   = lane >> 4;
  const int lid  = lane & 15;

  const int m0 = blockIdx.x * 256 + wave * 64;
  const int n0 = blockIdx.y * 64;

  const __bf16* arow[4];
#pragma unroll
  for (int a = 0; a < 4; ++a)
    arow[a] = aout + (size_t)(m0 + a * 16 + lid) * CDIM;

  const v8f vzero = {0.f, 0.f, 0.f, 0.f, 0.f, 0.f, 0.f, 0.f};
  v8f acc[4][4];
#pragma unroll
  for (int a = 0; a < 4; ++a)
#pragma unroll
    for (int j = 0; j < 4; ++j) acc[a][j] = vzero;

  for (int k = 0; k < CDIM; k += 32) {
    v16bf af[4];
#pragma unroll
    for (int a = 0; a < 4; ++a) af[a] = frag_a_bf16(arow[a] + k + 8 * hi);
#pragma unroll
    for (int j = 0; j < 4; ++j) {
      const __bf16* wrow = wb + (size_t)(n0 + j * 16 + lid) * CDIM;
      const v16bf b = frag_b_bf16(wrow + k + 8 * hi);
#pragma unroll
      for (int a = 0; a < 4; ++a) acc[a][j] = wmma_bf16(af[a], b, acc[a][j]);
    }
  }

#pragma unroll
  for (int a = 0; a < 4; ++a) {
#pragma unroll
    for (int j = 0; j < 4; ++j) {
      const int col = n0 + j * 16 + lid;
      const float bias = proj_b[col];
#pragma unroll
      for (int r = 0; r < 8; ++r)
        out[(size_t)(m0 + a * 16 + 8 * hi + r) * CDIM + col] = acc[a][j][r] + bias;
    }
  }
}

// ---------------------------------------------------------------------------
extern "C" void kernel_launch(void* const* d_in, const int* in_sizes, int n_in,
                              void* d_out, int out_size, void* d_ws, size_t ws_size,
                              hipStream_t stream) {
  const float* x      = (const float*)d_in[0];
  const float* mask   = (const float*)d_in[1];
  const float* qkv_w  = (const float*)d_in[2];
  const float* q_bias = (const float*)d_in[3];
  const float* v_bias = (const float*)d_in[4];
  const float* proj_w = (const float*)d_in[5];
  const float* proj_b = (const float*)d_in[6];
  float* out = (float*)d_out;

  const size_t QKV_ELEMS  = (size_t)BTH * NTOK * DH;   // 6,291,456
  const size_t X_ELEMS    = (size_t)MROWS * CDIM;      // 6,291,456
  const size_t WQKV_ELEMS = (size_t)3 * CDIM * CDIM;   // 1,769,472
  const size_t WPRJ_ELEMS = (size_t)CDIM * CDIM;       //   589,824

  __bf16* qws   = (__bf16*)d_ws;
  __bf16* kws   = qws + QKV_ELEMS;
  __bf16* vT    = kws + QKV_ELEMS;
  __bf16* aws   = vT + QKV_ELEMS;        // 8192*768 == QKV_ELEMS
  __bf16* xb    = aws + QKV_ELEMS;
  __bf16* wqkvb = xb + X_ELEMS;
  __bf16* wprjb = wqkvb + WQKV_ELEMS;
  // total workspace ~= (5*6291456 + 1769472 + 589824) * 2 bytes ~= 64.5 MiB

  // ---- prepass: convert x / qkv_w / proj_w to bf16 once (bandwidth-bound)
  {
    const int nx = (int)(X_ELEMS / 8);
    const int nw = (int)(WQKV_ELEMS / 8);
    const int np = (int)(WPRJ_ELEMS / 8);
    cvt_kernel<<<(nx + 255) / 256, 256, 0, stream>>>(x, xb, nx);
    cvt_kernel<<<(nw + 255) / 256, 256, 0, stream>>>(qkv_w, wqkvb, nw);
    cvt_kernel<<<(np + 255) / 256, 256, 0, stream>>>(proj_w, wprjb, np);
  }

  qkv_kernel<<<dim3(MROWS / 256, (3 * CDIM) / 64), 128, 0, stream>>>(
      xb, wqkvb, q_bias, v_bias, qws, kws, vT);
  attn_kernel<<<dim3(BTH, NTOK / 64), 128, 0, stream>>>(
      qws, kws, vT, mask, aws);
  proj_kernel<<<dim3(MROWS / 256, CDIM / 64), 128, 0, stream>>>(
      aws, wprjb, proj_b, out);
}
